// Attention_88888643158200
// MI455X (gfx1250) — matrix operand order repaired and checked by reading, not compiled, not run on hardware
//
#include <hip/hip_runtime.h>
#include <math.h>

// Problem constants (from reference): B=8, C=256, N=2048, scale = 1/sqrt(256)
#define BB 8
#define CC 256
#define NN 2048
#define SCALE 0.0625f

typedef __attribute__((ext_vector_type(16))) __bf16 v16bf;
typedef __attribute__((ext_vector_type(8)))  __bf16 v8bf;
typedef __attribute__((ext_vector_type(2)))  __bf16 v2bf;
typedef __attribute__((ext_vector_type(8)))  float  v8f;

__device__ inline v16bf bf16_combine(v8bf lo, v8bf hi) {
    v16bf r;
#pragma unroll
    for (int t = 0; t < 8; ++t) { r[t] = lo[t]; r[8 + t] = hi[t]; }
    return r;
}

// A-matrix fragment (16x32 bf16). `row` points at 32 contiguous bf16 K-values
// for this lane's M row. Per ISA: lanes 0-15 slots0-7 = K0-7, slots8-15 = K16-23;
// lanes 16-31 shifted by +8.  -> two 16B contiguous LDS loads per lane.
__device__ inline v16bf loadA16x32(const __bf16* row, int half) {
    const v8bf* p = (const v8bf*)row;
    return bf16_combine(p[half], p[2 + half]);
}

// B-matrix fragment (32x16 bf16). `row` points at 32 contiguous bf16 K-values
// for this lane's N column. Same K order as A: lanes 0-15 slots0-7 = K0-7,
// slots8-15 = K16-23; lanes 16-31 shifted by +8. -> two 16B contiguous loads.
__device__ inline v16bf loadB32x16(const __bf16* row, int half) {
    const v8bf* p = (const v8bf*)row;
    return bf16_combine(p[half], p[2 + half]);
}

// ---------------------------------------------------------------------------
// Kernel 1: logits[b,i,j] = SCALE * sum_c k[b,c,i] * q[b,c,j]
// Workgroup tile: 128(i) x 128(j), 8 waves, each wave 64x32 (4x2 WMMA tiles).
// ---------------------------------------------------------------------------
__global__ __launch_bounds__(256) void attn_gemm1(const float* __restrict__ q,
                                                  const float* __restrict__ k,
                                                  float* __restrict__ logits) {
    const int b  = blockIdx.z;
    const int i0 = blockIdx.y * 128;
    const int j0 = blockIdx.x * 128;

    __shared__ __bf16 kb[128][32]; // [i][c] : A rows contiguous in K(c)
    __shared__ __bf16 qb[128][32]; // [j][c] : B cols contiguous in K(c)

    const int tid  = threadIdx.x;
    const int w    = tid >> 5;
    const int lane = tid & 31;
    const int half = lane >> 4;
    const int l16  = lane & 15;
    const int wi   = (w >> 2) * 64; // wave i-offset in tile (0 or 64)
    const int wj   = (w & 3) * 32;  // wave j-offset in tile

    v8f acc[4][2] = {};

    const float* kbase = k + (size_t)b * CC * NN + i0;
    const float* qbase = q + (size_t)b * CC * NN + j0;

    // prefetch coordinates: 32 rows x 8 x 64B chunks covers a 32x128 f32 slab
    const int prow = tid >> 3;
    const int pcol = (tid & 7) * 16;

    for (int c0 = 0; c0 < CC; c0 += 32) {
        __syncthreads();
        // stage 32x128 slabs of K and Q as bf16; pack pairs along c so each
        // LDS write is a single ds_store_b32.
#pragma unroll
        for (int t = tid; t < 2048; t += 256) {
            int col = t & 127;
            int cc  = (t >> 7) * 2;
            const float* kp = &kbase[(size_t)(c0 + cc) * NN + col];
            const float* qp = &qbase[(size_t)(c0 + cc) * NN + col];
            v2bf kpk = { (__bf16)kp[0], (__bf16)kp[NN] };
            v2bf qpk = { (__bf16)qp[0], (__bf16)qp[NN] };
            *(v2bf*)&kb[col][cc] = kpk;
            *(v2bf*)&qb[col][cc] = qpk;
        }
        __syncthreads();

        // stream next slab toward L2/WGP$ while the WMMAs below execute
        if (c0 + 32 < CC) {
            __builtin_prefetch(&kbase[(size_t)(c0 + 32 + prow) * NN + pcol], 0, 1);
            __builtin_prefetch(&qbase[(size_t)(c0 + 32 + prow) * NN + pcol], 0, 1);
        }

        v16bf amat[4], bmat[2];
#pragma unroll
        for (int it = 0; it < 4; ++it)
            amat[it] = loadA16x32(&kb[wi + it * 16 + l16][0], half);
#pragma unroll
        for (int jt = 0; jt < 2; ++jt)
            bmat[jt] = loadB32x16(&qb[wj + jt * 16 + l16][0], half);
#pragma unroll
        for (int it = 0; it < 4; ++it)
#pragma unroll
            for (int jt = 0; jt < 2; ++jt)
                acc[it][jt] = __builtin_amdgcn_wmma_f32_16x16x32_bf16(
                    false, amat[it], false, bmat[jt], (short)0, acc[it][jt],
                    false, false);
    }

    // D layout: lane -> n = l16 (j), reg r -> m = r + half*8 (i)
#pragma unroll
    for (int it = 0; it < 4; ++it)
#pragma unroll
        for (int jt = 0; jt < 2; ++jt) {
            int jg = j0 + wj + jt * 16 + l16;
#pragma unroll
            for (int r = 0; r < 8; ++r) {
                int ig = i0 + wi + it * 16 + half * 8 + r;
                logits[(size_t)b * NN * NN + (size_t)ig * NN + jg] =
                    acc[it][jt][r] * SCALE;
            }
        }
}

// ---------------------------------------------------------------------------
// Kernel 2: per (b,j) online softmax stats over key axis i:
//   mx[b,j] = max_i logits, rs[b,j] = 1 / sum_i exp(logits - mx)
// ---------------------------------------------------------------------------
__global__ __launch_bounds__(256) void attn_stats(const float* __restrict__ logits,
                                                  float* __restrict__ mx,
                                                  float* __restrict__ rs) {
    const int b = blockIdx.y;
    const int j = blockIdx.x * 256 + threadIdx.x;
    const float* col = logits + (size_t)b * NN * NN + j;
    float m = -3.402823466e38f;
    float s = 0.0f;
    for (int i = 0; i < NN; ++i) {
        float x  = col[(size_t)i * NN];
        float mn = fmaxf(m, x);
        s = s * __expf(m - mn) + __expf(x - mn);
        m = mn;
    }
    mx[b * NN + j] = m;
    rs[b * NN + j] = 1.0f / s;
}

// ---------------------------------------------------------------------------
// Kernel 3: fused normalize + out GEMM.
// Workgroup owns (b, 64-query tile), all C=256 output rows.
//   w[b,i,j] = exp(logits - mx) * rs   (written back in place -> final weights)
//   out[b,c,j] = sum_i v[b,c,i] * w[b,i,j]   via bf16 WMMA over i.
// 8 waves: wave w owns c in [w*32, w*32+32) x all 64 j -> 2x4 WMMA tiles.
// ---------------------------------------------------------------------------
__global__ __launch_bounds__(256) void attn_gemm2(const float* __restrict__ v,
                                                  const float* __restrict__ mx,
                                                  const float* __restrict__ rs,
                                                  float* __restrict__ wgt,
                                                  float* __restrict__ outp) {
    const int b  = blockIdx.y;
    const int j0 = blockIdx.x * 64;

    __shared__ __bf16 vb[256][32]; // [c][i] : A rows contiguous in K(i)
    __shared__ __bf16 wb[64][32];  // [j][i] : B cols contiguous in K(i)

    const int tid  = threadIdx.x;
    const int w    = tid >> 5;
    const int lane = tid & 31;
    const int half = lane >> 4;
    const int l16  = lane & 15;
    const int wc   = w * 32; // wave's c offset

    // weights staging: thread owns column jc, even row base irb (pairs of rows)
    const int jc  = tid & 63;
    const int irb = (tid >> 6) * 2; // 0,2,4,6
    const float mxv = mx[b * NN + j0 + jc];
    const float rsv = rs[b * NN + j0 + jc];

    const float* vsrc = v + (size_t)b * CC * NN;
    float* wbase = wgt + (size_t)b * NN * NN;
    float* wcol  = wbase + j0 + jc;

    // prefetch coords for the next 32x64 logits tile (4 x 64B chunks per row)
    const int lpr = (tid >> 2) & 31;
    const int lpc = (tid & 3) * 16;

    v8f acc[2][4] = {};

    for (int i0 = 0; i0 < NN; i0 += 32) {
        __syncthreads();
        // stage V tile (256 x 32) as bf16: float2 global loads, packed stores
#pragma unroll
        for (int t = tid; t < 4096; t += 256) {
            int ip = t & 15, cc = t >> 4;
            const float2 vv = *(const float2*)&vsrc[(size_t)cc * NN + i0 + ip * 2];
            v2bf pk = { (__bf16)vv.x, (__bf16)vv.y };
            *(v2bf*)&vb[cc][ip * 2] = pk;
        }
        // stage + normalize + write-back weights tile (32 x 64), row pairs
#pragma unroll
        for (int n = 0; n < 4; ++n) {
            int ir = irb + n * 8; // even rows 0..30, pair (ir, ir+1)
            float lg0 = wcol[(size_t)(i0 + ir) * NN];
            float lg1 = wcol[(size_t)(i0 + ir + 1) * NN];
            float w0 = __expf(lg0 - mxv) * rsv;
            float w1 = __expf(lg1 - mxv) * rsv;
            wcol[(size_t)(i0 + ir) * NN]     = w0; // final normalized weight
            wcol[(size_t)(i0 + ir + 1) * NN] = w1;
            v2bf pk = { (__bf16)w0, (__bf16)w1 };
            *(v2bf*)&wb[jc][ir] = pk;
        }
        __syncthreads();

        if (i0 + 32 < NN) {
            __builtin_prefetch(&vsrc[(size_t)tid * NN + i0 + 32], 0, 1);
            __builtin_prefetch(&vsrc[(size_t)tid * NN + i0 + 48], 0, 1);
            __builtin_prefetch(&wbase[(size_t)(i0 + 32 + lpr) * NN + j0 + lpc], 0, 1);
        }

        v16bf amat[2], bmat[4];
#pragma unroll
        for (int ct = 0; ct < 2; ++ct)
            amat[ct] = loadA16x32(&vb[wc + ct * 16 + l16][0], half);
#pragma unroll
        for (int jt = 0; jt < 4; ++jt)
            bmat[jt] = loadB32x16(&wb[jt * 16 + l16][0], half);
#pragma unroll
        for (int ct = 0; ct < 2; ++ct)
#pragma unroll
            for (int jt = 0; jt < 4; ++jt)
                acc[ct][jt] = __builtin_amdgcn_wmma_f32_16x16x32_bf16(
                    false, amat[ct], false, bmat[jt], (short)0, acc[ct][jt],
                    false, false);
    }

#pragma unroll
    for (int ct = 0; ct < 2; ++ct)
#pragma unroll
        for (int jt = 0; jt < 4; ++jt) {
            int jg = j0 + jt * 16 + l16;
#pragma unroll
            for (int r = 0; r < 8; ++r) {
                int cg = wc + ct * 16 + half * 8 + r;
                outp[(size_t)b * CC * NN + (size_t)cg * NN + jg] = acc[ct][jt][r];
            }
        }
}

extern "C" void kernel_launch(void* const* d_in, const int* in_sizes, int n_in,
                              void* d_out, int out_size, void* d_ws, size_t ws_size,
                              hipStream_t stream) {
    (void)in_sizes; (void)n_in; (void)out_size; (void)ws_size;
    const float* q = (const float*)d_in[0];
    const float* k = (const float*)d_in[1];
    const float* v = (const float*)d_in[2];

    float* outp = (float*)d_out;                       // [B, C, N]
    float* wgt  = outp + (size_t)BB * CC * NN;         // [B, N, N] (logits -> weights)

    float* mx = (float*)d_ws;                          // [B, N]
    float* rs = mx + (size_t)BB * NN;                  // [B, N]

    attn_gemm1<<<dim3(NN / 128, NN / 128, BB), 256, 0, stream>>>(q, k, wgt);
    attn_stats<<<dim3(NN / 256, BB), 256, 0, stream>>>(wgt, mx, rs);
    attn_gemm2<<<dim3(NN / 64, BB), 256, 0, stream>>>(v, mx, rs, wgt, outp);
}
